// minerva_56135222559424
// MI455X (gfx1250) — matrix operand order repaired and checked by reading, not compiled, not run on hardware
//
#include <hip/hip_runtime.h>
#include <hip/hip_bf16.h>
#include <math.h>

// ---------------------------------------------------------------------------
// Problem constants (match reference)
// ---------------------------------------------------------------------------
#define BATCH   1024
#define D_IN    1024
#define D_FEAT  256
#define N_EX    100000
#define N_LAB   28
#define N_LABP  32          // labels padded to 32 for WMMA N-tiles
#define CHUNK   32          // exemplars per fused-kernel chunk (100000 % 32 == 0)
#define EPS     1e-12f

typedef __attribute__((ext_vector_type(16))) __bf16        v16bf;
typedef __attribute__((ext_vector_type(8)))  float         v8f;
typedef __attribute__((ext_vector_type(4)))  float         f4;
typedef __attribute__((ext_vector_type(4)))  unsigned int  u4;

union Frag { u4 q[2]; unsigned int u[8]; v16bf v; };

// ---------------------------------------------------------------------------
// CDNA5 async global->LDS staging (ASYNCcnt-tracked), inline asm per
// CDNA5 ISA. GVS mode: mem = SADDR64 + VADDR32 + IOFFSET.
// VDST holds the LDS byte address.
// ---------------------------------------------------------------------------
__device__ __forceinline__ void async_b128(unsigned lds_addr, unsigned goff,
                                           unsigned long long gbase) {
    asm volatile("global_load_async_to_lds_b128 %0, %1, %2"
                 :: "v"(lds_addr), "v"(goff), "s"(gbase)
                 : "memory");
}

__device__ __forceinline__ void wait_async0() {
#if defined(__has_builtin)
#  if __has_builtin(__builtin_amdgcn_s_wait_asynccnt)
    __builtin_amdgcn_s_wait_asynccnt(0);
#  else
    asm volatile("s_wait_asynccnt 0x0" ::: "memory");
#  endif
#else
    asm volatile("s_wait_asynccnt 0x0" ::: "memory");
#endif
}

// Stage one contiguous 16 KB chunk (32 exemplar rows x 512 B) into LDS.
// 256 threads x 4 x B128 = 16 KB.
__device__ __forceinline__ void stage_chunk(const __bf16* __restrict__ gsrc,
                                            unsigned ldsBase, int tid) {
    unsigned long long g = (unsigned long long)(size_t)gsrc;
#pragma unroll
    for (int i = 0; i < 4; ++i) {
        unsigned off = (unsigned)(tid * 16 + i * 4096);
        async_b128(ldsBase + off, off, g);
    }
}

// ---------------------------------------------------------------------------
// Kernel: zero a float buffer
// ---------------------------------------------------------------------------
__global__ void zero_f32_kernel(float* __restrict__ p, int n) {
    int i = blockIdx.x * blockDim.x + threadIdx.x;
    if (i < n) p[i] = 0.0f;
}

// ---------------------------------------------------------------------------
// Kernel: C = l2norm_rows(A[rows,1024] @ W[256,1024]^T) stored as bf16.
// 256 threads = 8 waves; block = 16 rows x 256 cols; wave w covers cols
// [32w, 32w+32) via two 16x16x32 bf16 WMMA tiles. rows % 16 == 0.
// A is streamed once -> nontemporal loads (don't thrash L2); W stays RT/hot.
// ---------------------------------------------------------------------------
__global__ void proj_norm_kernel(const float* __restrict__ A,
                                 const float* __restrict__ W,
                                 __bf16* __restrict__ out,
                                 int rows) {
    __shared__ float rowsq[16];
    const int tid  = threadIdx.x;
    const int lane = tid & 31;
    const int w    = tid >> 5;
    const int half = lane >> 4;
    const int ln   = lane & 15;
    const int rowBase = blockIdx.x * 16;
    const int n0   = w * 32;

    v8f c0 = {}; v8f c1 = {};

    const float* pArow  = A + (size_t)(rowBase + ln) * D_IN;
    const float* pB0row = W + (size_t)(n0 + ln) * D_IN;
    const float* pB1row = W + (size_t)(n0 + 16 + ln) * D_IN;

    for (int kb = 0; kb < D_IN; kb += 32) {
        // A-frag floats: [kb+16h .. +16)                       (contiguous, matches B)
        const f4* pa0 = (const f4*)(pArow + kb + 16 * half);
        const f4* pa2 = (const f4*)(pArow + kb + 16 * half + 8);
        f4 a0 = __builtin_nontemporal_load(pa0);
        f4 a1 = __builtin_nontemporal_load(pa0 + 1);
        f4 a2 = __builtin_nontemporal_load(pa2);
        f4 a3 = __builtin_nontemporal_load(pa2 + 1);
        // B-frag floats: [kb+16h .. +16)                       (contiguous, matches A)
        const f4* pb0 = (const f4*)(pB0row + kb + 16 * half);
        const f4* pb1 = (const f4*)(pB1row + kb + 16 * half);
        f4 b00 = pb0[0], b01 = pb0[1], b02 = pb0[2], b03 = pb0[3];
        f4 b10 = pb1[0], b11 = pb1[1], b12 = pb1[2], b13 = pb1[3];

        v16bf av, b0v, b1v;
#pragma unroll
        for (int i = 0; i < 4; ++i) {
            av[i]      = (__bf16)a0[i];
            av[4 + i]  = (__bf16)a1[i];
            av[8 + i]  = (__bf16)a2[i];
            av[12 + i] = (__bf16)a3[i];
            b0v[i]      = (__bf16)b00[i];
            b0v[4 + i]  = (__bf16)b01[i];
            b0v[8 + i]  = (__bf16)b02[i];
            b0v[12 + i] = (__bf16)b03[i];
            b1v[i]      = (__bf16)b10[i];
            b1v[4 + i]  = (__bf16)b11[i];
            b1v[8 + i]  = (__bf16)b12[i];
            b1v[12 + i] = (__bf16)b13[i];
        }
        c0 = __builtin_amdgcn_wmma_f32_16x16x32_bf16(false, av, false, b0v,
                                                     (short)0, c0, false, false);
        c1 = __builtin_amdgcn_wmma_f32_16x16x32_bf16(false, av, false, b1v,
                                                     (short)0, c1, false, false);
    }

    if (tid < 16) rowsq[tid] = 0.0f;
    __syncthreads();
#pragma unroll
    for (int r = 0; r < 8; ++r)
        atomicAdd(&rowsq[r + 8 * half], c0[r] * c0[r] + c1[r] * c1[r]);
    __syncthreads();

#pragma unroll
    for (int r = 0; r < 8; ++r) {
        const int m = r + 8 * half;
        float inv = 1.0f / fmaxf(sqrtf(rowsq[m]), EPS);
        __bf16* po = out + (size_t)(rowBase + m) * D_FEAT;
        po[n0 + ln]      = (__bf16)(c0[r] * inv);
        po[n0 + 16 + ln] = (__bf16)(c1[r] * inv);
    }
}

// ---------------------------------------------------------------------------
// Kernel: ecnT[c][e] = l2norm_row(ex_classes[e,:])[c] as bf16, rows 28..31 = 0
// ---------------------------------------------------------------------------
__global__ void ecn_transpose_kernel(const float* __restrict__ exc,
                                     __bf16* __restrict__ ecnT) {
    int e = blockIdx.x * blockDim.x + threadIdx.x;
    if (e >= N_EX) return;
    float v[N_LAB];
    float s = 0.0f;
#pragma unroll
    for (int c = 0; c < N_LAB; ++c) {
        v[c] = exc[(size_t)e * N_LAB + c];
        s += v[c] * v[c];
    }
    float inv = 1.0f / fmaxf(sqrtf(s), EPS);
#pragma unroll
    for (int c = 0; c < N_LAB; ++c)
        ecnT[(size_t)c * N_EX + e] = (__bf16)(v[c] * inv);
#pragma unroll
    for (int c = N_LAB; c < N_LABP; ++c)
        ecnT[(size_t)c * N_EX + e] = (__bf16)0.0f;
}

// ---------------------------------------------------------------------------
// Fused Minerva kernel (64 batch rows per workgroup, async double-buffered):
//   s = fn @ efn^T (bf16 WMMA, K=256), a = s^3, echo += a @ ecn (bf16 WMMA)
// Block: 256 threads = 8 waves; wave w: row-tile rt = w>>1, exemplar tile
// et = w&1 (s phase), label half nh = w&1 (echo phase). Chunk = 32 exemplars,
// staged into LDS with GLOBAL_LOAD_ASYNC_TO_LDS_B128 (prefetch next chunk
// while computing current). 100000 % 32 == 0 -> no tail guards anywhere.
// ---------------------------------------------------------------------------
__global__ void minerva_fused_kernel(const __bf16* __restrict__ fn,    // [B, 256]
                                     const __bf16* __restrict__ efn,   // [N_EX, 256]
                                     const __bf16* __restrict__ ecnT,  // [32, N_EX]
                                     float* __restrict__ echo) {       // [B, 28]
    __shared__ __bf16 efnLds[2][CHUNK * D_FEAT];   // 2 x 16 KB
    __shared__ __bf16 aLds[64 * CHUNK];            // 4 KB: a (cubed sims), bf16

    const int tid  = threadIdx.x;
    const int lane = tid & 31;
    const int w    = tid >> 5;
    const int half = lane >> 4;
    const int ln   = lane & 15;
    const int rt   = w >> 1;           // row tile 0..3
    const int et   = w & 1;            // exemplar tile (s phase)
    const int nh   = w & 1;            // label half   (echo phase)
    const int rowBase = blockIdx.x * 64 + rt * 16;

    const unsigned ldsBuf[2] = {
        (unsigned)(size_t)(void*)&efnLds[0][0],
        (unsigned)(size_t)(void*)&efnLds[1][0]
    };

    // Preload this wave's A fragments of fn (16 rows x K=256): 8 ksteps x 2 u4
    u4 afr[8][2];
    {
        const u4* pA = (const u4*)(fn + (size_t)(rowBase + ln) * D_FEAT);
#pragma unroll
        for (int t = 0; t < 8; ++t) {
            afr[t][0] = pA[t * 4 + half * 2];
            afr[t][1] = pA[t * 4 + half * 2 + 1];
        }
    }

    v8f eacc = {};
    const int NCHUNK = N_EX / CHUNK;   // 3125 exact

    int ch = blockIdx.y;
    int buf = 0;
    stage_chunk(efn + (size_t)ch * CHUNK * D_FEAT, ldsBuf[0], tid);

    for (; ch < NCHUNK; ch += gridDim.y) {
        wait_async0();
        __syncthreads();               // efnLds[buf] ready; prev aLds reads done

        const int nch = ch + gridDim.y;
        if (nch < NCHUNK)              // uniform branch (EXEC stays full)
            stage_chunk(efn + (size_t)nch * CHUNK * D_FEAT, ldsBuf[buf ^ 1], tid);

        // ---- s tile: rows [rowBase,+16) x exemplars [ch*32+et*16,+16) ----
        v8f sacc = {};
        const __bf16* bRow = &efnLds[buf][(et * 16 + ln) * D_FEAT];
#pragma unroll
        for (int t = 0; t < 8; ++t) {
            Frag au, bu;
            au.q[0] = afr[t][0];
            au.q[1] = afr[t][1];
            const u4* pB = (const u4*)(bRow + t * 32);
            bu.q[0] = pB[half * 2];
            bu.q[1] = pB[half * 2 + 1];
            sacc = __builtin_amdgcn_wmma_f32_16x16x32_bf16(false, au.v, false, bu.v,
                                                           (short)0, sacc, false, false);
        }

        // ---- a = s^3 (sign(s)|s|^3, p=3 odd), restage bf16 into LDS ----
#pragma unroll
        for (int r = 0; r < 8; ++r) {
            float s = sacc[r];
            aLds[(rt * 16 + r + 8 * half) * CHUNK + et * 16 + ln] = (__bf16)(s * s * s);
        }
        __syncthreads();               // aLds visible to echo phase

        // ---- echo WMMA: D[16 x 16 labels] += a[16 x 32ex] @ ecn[32ex x 16] ----
        {
            Frag ea, eb;
            const u4* pa = (const u4*)(aLds + (rt * 16 + ln) * CHUNK);
            ea.q[0] = pa[half * 2];
            ea.q[1] = pa[half * 2 + 1];
            const u4* pe = (const u4*)(ecnT + (size_t)(nh * 16 + ln) * N_EX
                                       + (size_t)ch * CHUNK);
            eb.q[0] = pe[half * 2];
            eb.q[1] = pe[half * 2 + 1];
            eacc = __builtin_amdgcn_wmma_f32_16x16x32_bf16(false, ea.v, false, eb.v,
                                                           (short)0, eacc, false, false);
        }
        buf ^= 1;
    }

    // ---- commit: each (rt, nh) owned by exactly one wave -> direct atomics ----
#pragma unroll
    for (int r = 0; r < 8; ++r) {
        const int m = r + 8 * half;
        const int c = nh * 16 + ln;
        if (c < N_LAB)
            atomicAdd(&echo[(size_t)(rowBase + m) * N_LAB + c], eacc[r]);
    }
}

// ---------------------------------------------------------------------------
// logits[b][j] = -sqrt(||echo_b||^2 - 2*echo_b[j] + 1)   (class_reps = I)
// ---------------------------------------------------------------------------
__global__ void logits_kernel(const float* __restrict__ echo,
                              float* __restrict__ out) {
    const int b = blockIdx.x;
    const int c = threadIdx.x;                   // 0..31
    float v = (c < N_LAB) ? echo[(size_t)b * N_LAB + c] : 0.0f;
    float s = v * v;
#pragma unroll
    for (int off = 16; off > 0; off >>= 1) s += __shfl_xor(s, off, 32);
    if (c < N_LAB)
        out[(size_t)b * N_LAB + c] = -sqrtf(fmaxf(s - 2.0f * v + 1.0f, 0.0f));
}

// ---------------------------------------------------------------------------
// Host launcher
// ---------------------------------------------------------------------------
extern "C" void kernel_launch(void* const* d_in, const int* in_sizes, int n_in,
                              void* d_out, int out_size, void* d_ws, size_t ws_size,
                              hipStream_t stream) {
    const float* features    = (const float*)d_in[0];   // [1024, 1024]
    const float* ex_features = (const float*)d_in[1];   // [100000, 1024]
    const float* ex_classes  = (const float*)d_in[2];   // [100000, 28]
    const float* g_w         = (const float*)d_in[3];   // [256, 1024]
    // d_in[4] = class_reps (identity) -- folded analytically, unused.

    char* ws = (char*)d_ws;
    size_t off = 0;
    __bf16* efn  = (__bf16*)(ws + off); off += (size_t)N_EX * D_FEAT * 2;   // 51.2 MB
    __bf16* ecnT = (__bf16*)(ws + off); off += (size_t)N_LABP * N_EX * 2;   //  6.4 MB
    __bf16* fn   = (__bf16*)(ws + off); off += (size_t)BATCH * D_FEAT * 2;  //  0.5 MB
    float*  echo = (float*)(ws + off);  off += (size_t)BATCH * N_LAB * 4;   //  0.1 MB

    zero_f32_kernel<<<(BATCH * N_LAB + 255) / 256, 256, 0, stream>>>(echo,
                                                                     BATCH * N_LAB);
    proj_norm_kernel<<<BATCH / 16, 256, 0, stream>>>(features, g_w, fn, BATCH);
    proj_norm_kernel<<<N_EX / 16, 256, 0, stream>>>(ex_features, g_w, efn, N_EX);
    ecn_transpose_kernel<<<(N_EX + 255) / 256, 256, 0, stream>>>(ex_classes, ecnT);

    dim3 grid(BATCH / 64, 392);   // 16 row blocks x 392 chunk slices = 6272 WGs
    minerva_fused_kernel<<<grid, 256, 0, stream>>>(fn, efn, ecnT, echo);

    logits_kernel<<<BATCH, 32, 0, stream>>>(echo, (float*)d_out);
}
